// TokenKANBlock_33122787787481
// MI455X (gfx1250) — matrix operand order repaired and checked by reading, not compiled, not run on hardware
//
#include <hip/hip_runtime.h>

// ---------------------------------------------------------------------------
// Types for CDNA5 WMMA (wave32): v_wmma_f32_16x16x32_bf16
// ---------------------------------------------------------------------------
typedef __bf16 bf16_t;
typedef __attribute__((ext_vector_type(16))) __bf16 v16bf;
typedef __attribute__((ext_vector_type(8)))  float  v8f;

__device__ __forceinline__ bf16_t f2bf(float f) {
  unsigned u = __builtin_bit_cast(unsigned, f);
  u += 0x7fffu + ((u >> 16) & 1u);               // round-to-nearest-even
  unsigned short h = (unsigned short)(u >> 16);
  return __builtin_bit_cast(bf16_t, h);
}

// Low 32 bits of a generic pointer to LDS == LDS byte offset (AMDGPU
// addrspacecast AS3->flat places the 32-bit LDS pointer in the low dword).
__device__ __forceinline__ unsigned lds_addr32(const void* p) {
  return (unsigned)(unsigned long long)p;
}

// Async global->LDS copy of 16 bytes (GLOBAL_LOAD_ASYNC_TO_LDS_B128,
// tracked by ASYNCcnt). Data bypasses VGPRs entirely.
__device__ __forceinline__ void async_b128(const void* gaddr, void* laddr) {
  unsigned lds = lds_addr32(laddr);
  unsigned long long ga = (unsigned long long)gaddr;
  asm volatile("global_load_async_to_lds_b128 %0, %1, off"
               :: "v"(lds), "v"(ga) : "memory");
}
__device__ __forceinline__ void wait_async_le8() {
  asm volatile("s_wait_asynccnt 0x8" ::: "memory");
}
__device__ __forceinline__ void wait_async_0() {
  asm volatile("s_wait_asynccnt 0x0" ::: "memory");
}

// ---------------------------------------------------------------------------
// WMMA fragment loaders (wave32); A and B use contiguous k = 16*half + e, not the ISA table order.
// A (16x32 bf16, MxK): lane<16 holds row M=lane, K in {0..15};
//                      lane>=16 holds row M=lane-16, K in {16..31}.
// B (32x16 bf16, KxN): element e -> K = half*16+e, N = lane&15.
// C/D (16x16 f32):     vgpr r -> M = r + 8*half, N = lane&15.
// ---------------------------------------------------------------------------
__device__ __forceinline__ v16bf load_a_frag(const bf16_t* s, int ld) {
  int lane = threadIdx.x & 31;
  int hf = lane >> 4;
  int m  = lane & 15;
  const bf16_t* row = s + (size_t)m * ld;
  v16bf a;
#pragma unroll
  for (int v = 0; v < 8; ++v) {
    int kb = hf * 16 + v * 2;
    a[2 * v]     = row[kb];
    a[2 * v + 1] = row[kb + 1];
  }
  return a;
}

// B source stored as (N rows, K cols) row-major (weight layout (out, in)).
__device__ __forceinline__ v16bf load_b_frag_nk(const bf16_t* s, int ld) {
  int lane = threadIdx.x & 31;
  int hf = lane >> 4;
  int n  = lane & 15;
  const bf16_t* p = s + (size_t)n * ld + hf * 16;
  v16bf b;
#pragma unroll
  for (int e = 0; e < 16; ++e) b[e] = p[e];
  return b;
}

// B source stored as (K rows, N cols) row-major (e.g. V tile in attention).
__device__ __forceinline__ v16bf load_b_frag_kn(const bf16_t* s, int ld) {
  int lane = threadIdx.x & 31;
  int hf = lane >> 4;
  int n  = lane & 15;
  v16bf b;
#pragma unroll
  for (int e = 0; e < 16; ++e) b[e] = s[(size_t)(hf * 16 + e) * ld + n];
  return b;
}

// ---------------------------------------------------------------------------
// Generic bf16 GEMM: C[M,N](f32) = A[M,K](bf16,row-major) * B[N,K](bf16)^T
//                                  (+ bias[N] if bias != nullptr)
// Tile 128x128x64, 256 threads = 8 waves (2x4), each wave: 64x32 output,
// 16 WMMA per K-tile. Tiles streamed with double-buffered
// GLOBAL_LOAD_ASYNC_TO_LDS_B128 (ASYNCcnt pipelined one tile ahead).
// Requires M%128==0, N%128==0, K%64==0 (true for all call sites here).
// ---------------------------------------------------------------------------
#define BK 64
__global__ __launch_bounds__(256) void k_gemm_bf16(
    const bf16_t* __restrict__ A, const bf16_t* __restrict__ B,
    const float* __restrict__ bias, float* __restrict__ C,
    int M, int N, int K) {
  __shared__ __align__(16) bf16_t sA[2][128 * BK];
  __shared__ __align__(16) bf16_t sB[2][128 * BK];

  const int t  = threadIdx.x;
  const int bx = blockIdx.x, by = blockIdx.y;
  const int wv = t >> 5;
  const int wm = wv >> 2;      // 0..1
  const int wn = wv & 3;       // 0..3

  const bf16_t* Ab = A + (size_t)(by * 128) * K;
  const bf16_t* Bb = B + (size_t)(bx * 128) * K;

  const int r  = t >> 1;           // tile row 0..127
  const int cc = (t & 1) * 32;     // tile col offset 0 / 32

  // Issue 4+4 async b128 loads: one K-slice of the A and B tiles.
  auto issue_tile = [&](int k0, int p) {
#pragma unroll
    for (int j = 0; j < 4; ++j) {
      async_b128(Ab + (size_t)r * K + k0 + cc + j * 8,
                 &sA[p][r * BK + cc + j * 8]);
    }
#pragma unroll
    for (int j = 0; j < 4; ++j) {
      async_b128(Bb + (size_t)r * K + k0 + cc + j * 8,
                 &sB[p][r * BK + cc + j * 8]);
    }
  };

  v8f acc[4][2] = {};

  issue_tile(0, 0);                 // prologue: tile 0 in flight
  int p = 0;
  for (int k0 = 0; k0 < K; k0 += BK, p ^= 1) {
    if (k0 + BK < K) {
      issue_tile(k0 + BK, p ^ 1);   // prefetch next tile into other buffer
      wait_async_le8();             // previous 8 (this tile) have completed
    } else {
      wait_async_0();
    }
    __syncthreads();                // tile visible to all waves

#pragma unroll
    for (int ks = 0; ks < 2; ++ks) {
      v16bf bf0 = load_b_frag_nk(sB[p] + (wn * 32 + 0)  * BK + ks * 32, BK);
      v16bf bf1 = load_b_frag_nk(sB[p] + (wn * 32 + 16) * BK + ks * 32, BK);
#pragma unroll
      for (int mt = 0; mt < 4; ++mt) {
        v16bf af = load_a_frag(sA[p] + (wm * 64 + mt * 16) * BK + ks * 32, BK);
        acc[mt][0] = __builtin_amdgcn_wmma_f32_16x16x32_bf16(
            false, af, false, bf0, (short)0, acc[mt][0], false, false);
        acc[mt][1] = __builtin_amdgcn_wmma_f32_16x16x32_bf16(
            false, af, false, bf1, (short)0, acc[mt][1], false, false);
      }
    }
    __syncthreads();                // all waves done with buffer p
  }

  const int lane = t & 31, hf = lane >> 4, n = lane & 15;
#pragma unroll
  for (int mt = 0; mt < 4; ++mt) {
#pragma unroll
    for (int nt = 0; nt < 2; ++nt) {
      int r0 = by * 128 + wm * 64 + mt * 16 + 8 * hf;
      int c0 = bx * 128 + wn * 32 + nt * 16 + n;
      float bv = bias ? bias[c0] : 0.0f;
      float* outp = C + (size_t)r0 * N + c0;
#pragma unroll
      for (int rr = 0; rr < 8; ++rr) outp[(size_t)rr * N] = acc[mt][nt][rr] + bv;
    }
  }
}

// ---------------------------------------------------------------------------
// Flash-style attention. One workgroup = (b, h, 64 q-rows), 4 waves, each wave
// owns 16 q rows. Q/K/V are bf16 in (B,H,S,64). Output AO is bf16 in (B,S,E)
// with column = h*64+d (ready to feed the out-projection GEMM).
// K/V tiles streamed with double-buffered async loads.
// ---------------------------------------------------------------------------
__global__ __launch_bounds__(128) void k_attn(
    const bf16_t* __restrict__ Q, const bf16_t* __restrict__ Kt,
    const bf16_t* __restrict__ V, bf16_t* __restrict__ AO,
    int S, int H) {
  __shared__ __align__(16) bf16_t sK[2][64 * 64];
  __shared__ __align__(16) bf16_t sV[2][64 * 64];
  __shared__ __align__(16) bf16_t sP[4][16 * 64];

  const int t = threadIdx.x;
  const int w = t >> 5;
  const int lane = t & 31, hf = lane >> 4, n = lane & 15;
  const int q0 = blockIdx.x * 64;
  const int h  = blockIdx.y;
  const int b  = blockIdx.z;
  const int E  = H * 64;

  const bf16_t* qbase = Q + ((size_t)(b * H + h) * S + q0 + w * 16) * 64;
  const bf16_t* kbase = Kt + (size_t)(b * H + h) * S * 64;
  const bf16_t* vbase = V + (size_t)(b * H + h) * S * 64;

  v16bf aq0 = load_a_frag(qbase, 64);
  v16bf aq1 = load_a_frag(qbase + 32, 64);

  // 128 threads: each covers row t>>1, cols (t&1)*32..+32 of the 64x64 tiles.
  const int lr = t >> 1;
  const int lc = (t & 1) * 32;
  auto issue_kv = [&](int kv, int p) {
#pragma unroll
    for (int j = 0; j < 4; ++j)
      async_b128(kbase + (size_t)(kv + lr) * 64 + lc + j * 8,
                 &sK[p][lr * 64 + lc + j * 8]);
#pragma unroll
    for (int j = 0; j < 4; ++j)
      async_b128(vbase + (size_t)(kv + lr) * 64 + lc + j * 8,
                 &sV[p][lr * 64 + lc + j * 8]);
  };

  v8f o[4] = {};
  float mrow[8], lrow[8];
#pragma unroll
  for (int rr = 0; rr < 8; ++rr) { mrow[rr] = -1e30f; lrow[rr] = 0.0f; }

  issue_kv(0, 0);
  int p = 0;
  for (int kv0 = 0; kv0 < S; kv0 += 64, p ^= 1) {
    if (kv0 + 64 < S) {
      issue_kv(kv0 + 64, p ^ 1);
      wait_async_le8();
    } else {
      wait_async_0();
    }
    __syncthreads();

    // S = (Q K^T) / sqrt(64): four 16x16 tiles over 64 kv columns
    v8f st[4];
#pragma unroll
    for (int nt = 0; nt < 4; ++nt) {
      v8f c = {};
      v16bf bk0 = load_b_frag_nk(sK[p] + (size_t)(nt * 16) * 64, 64);
      v16bf bk1 = load_b_frag_nk(sK[p] + (size_t)(nt * 16) * 64 + 32, 64);
      c = __builtin_amdgcn_wmma_f32_16x16x32_bf16(false, aq0, false, bk0,
                                                  (short)0, c, false, false);
      c = __builtin_amdgcn_wmma_f32_16x16x32_bf16(false, aq1, false, bk1,
                                                  (short)0, c, false, false);
      st[nt] = c * 0.125f;
    }

    // Online softmax (row reductions across 16-lane halves via shuffles)
    float mnew[8], sc[8];
#pragma unroll
    for (int rr = 0; rr < 8; ++rr) {
      float v = fmaxf(fmaxf(st[0][rr], st[1][rr]), fmaxf(st[2][rr], st[3][rr]));
#pragma unroll
      for (int off = 8; off >= 1; off >>= 1)
        v = fmaxf(v, __shfl_xor(v, off, 32));
      mnew[rr] = fmaxf(mrow[rr], v);
      sc[rr]   = __expf(mrow[rr] - mnew[rr]);
      mrow[rr] = mnew[rr];
    }

    bf16_t* sp = sP[w];
#pragma unroll
    for (int rr = 0; rr < 8; ++rr) {
      float psum = 0.0f;
#pragma unroll
      for (int nt = 0; nt < 4; ++nt) {
        float pe = __expf(st[nt][rr] - mnew[rr]);
        psum += pe;
        sp[(size_t)(rr + 8 * hf) * 64 + nt * 16 + n] = f2bf(pe);
      }
#pragma unroll
      for (int off = 8; off >= 1; off >>= 1) psum += __shfl_xor(psum, off, 32);
      lrow[rr] = lrow[rr] * sc[rr] + psum;
    }
#pragma unroll
    for (int nt = 0; nt < 4; ++nt)
#pragma unroll
      for (int rr = 0; rr < 8; ++rr) o[nt][rr] *= sc[rr];

    // O += P @ V  (P re-layouted through per-wave LDS buffer)
#pragma unroll
    for (int ks = 0; ks < 2; ++ks) {
      v16bf pf = load_a_frag(sp + ks * 32, 64);
#pragma unroll
      for (int nt = 0; nt < 4; ++nt) {
        v16bf bv = load_b_frag_kn(sV[p] + (size_t)(ks * 32) * 64 + nt * 16, 64);
        o[nt] = __builtin_amdgcn_wmma_f32_16x16x32_bf16(
            false, pf, false, bv, (short)0, o[nt], false, false);
      }
    }
    __syncthreads();   // all waves done with buffer p before it is refilled
  }

  bf16_t* aob = AO + ((size_t)b * S + q0 + w * 16 + 8 * hf) * E + h * 64;
#pragma unroll
  for (int nt = 0; nt < 4; ++nt)
#pragma unroll
    for (int rr = 0; rr < 8; ++rr)
      aob[(size_t)rr * E + nt * 16 + n] = f2bf(o[nt][rr] / lrow[rr]);
}

// ---------------------------------------------------------------------------
// Elementwise / helper kernels
// ---------------------------------------------------------------------------
__global__ void k_cvt_bf16(const float* __restrict__ in,
                           bf16_t* __restrict__ out, size_t nelem) {
  size_t i = (size_t)blockIdx.x * blockDim.x + threadIdx.x;
  if (i < nelem) out[i] = f2bf(in[i]);
}

// qkv (N, 3E) f32 -> q/k/v bf16 in (B,H,S,64)
__global__ void k_split_qkv(const float* __restrict__ qkv,
                            bf16_t* __restrict__ q, bf16_t* __restrict__ k,
                            bf16_t* __restrict__ v, int B, int H, int S) {
  size_t idx = (size_t)blockIdx.x * blockDim.x + threadIdx.x;
  size_t total = (size_t)B * H * S * 64;
  if (idx >= total) return;
  int d = idx & 63;
  size_t t2 = idx >> 6;
  int s = t2 % S; t2 /= S;
  int h = t2 % H;
  int b = (int)(t2 / H);
  int E = H * 64;
  const float* src = qkv + ((size_t)b * S + s) * (3 * E) + h * 64 + d;
  q[idx] = f2bf(src[0]);
  k[idx] = f2bf(src[E]);
  v[idx] = f2bf(src[2 * E]);
}

// out = LayerNorm(a + b) * scale + bias, optionally * sigmoid(terrain[b,:])
__global__ __launch_bounds__(256) void k_add_ln(
    const float* __restrict__ a, const float* __restrict__ b2,
    const float* __restrict__ scale, const float* __restrict__ bias,
    const float* __restrict__ terrain, float* __restrict__ out,
    int E, int S) {
  int row = blockIdx.x;
  int t = threadIdx.x;
  __shared__ float red[256];
  float vals[3];
  float s1 = 0.0f;
#pragma unroll
  for (int c = 0; c < 3; ++c) {
    int j = c * 256 + t;
    float v = a[(size_t)row * E + j] + b2[(size_t)row * E + j];
    vals[c] = v;
    s1 += v;
  }
  red[t] = s1; __syncthreads();
  for (int o = 128; o > 0; o >>= 1) {
    if (t < o) red[t] += red[t + o];
    __syncthreads();
  }
  float mean = red[0] / (float)E;
  __syncthreads();
  float s2 = 0.0f;
#pragma unroll
  for (int c = 0; c < 3; ++c) { float d = vals[c] - mean; s2 += d * d; }
  red[t] = s2; __syncthreads();
  for (int o = 128; o > 0; o >>= 1) {
    if (t < o) red[t] += red[t + o];
    __syncthreads();
  }
  float inv = rsqrtf(red[0] / (float)E + 1e-5f);
  int bidx = row / S;
#pragma unroll
  for (int c = 0; c < 3; ++c) {
    int j = c * 256 + t;
    float y = (vals[c] - mean) * inv * scale[j] + bias[j];
    if (terrain) {
      float tv = terrain[(size_t)bidx * E + j];
      y *= 1.0f / (1.0f + __expf(-tv));
    }
    out[(size_t)row * E + j] = y;
  }
}

// Build augmented KAN activation row: [silu(h_i) | bases(h_i,k=0..7)] (bf16).
// Cubic B-splines on the uniform grid of the reference (12 knots, h=0.4).
__global__ void k_kan_prep(const float* __restrict__ h,
                           bf16_t* __restrict__ aug, int I, size_t rows) {
  size_t idx = (size_t)blockIdx.x * blockDim.x + threadIdx.x;
  if (idx >= rows * (size_t)I) return;
  int i = (int)(idx % I);
  size_t nrow = idx / I;
  float x = h[idx];
  bf16_t* outr = aug + nrow * (size_t)(I * 9);
  outr[i] = f2bf(x / (1.0f + __expf(-x)));  // silu

  float tg[12];
#pragma unroll
  for (int g = 0; g < 12; ++g) tg[g] = (float)(g - 3) * 0.4f - 1.0f;
  float bb[11];
#pragma unroll
  for (int j = 0; j < 11; ++j)
    bb[j] = (x >= tg[j] && x < tg[j + 1]) ? 1.0f : 0.0f;
#pragma unroll
  for (int k = 1; k <= 3; ++k) {
    for (int j = 0; j < 11 - k; ++j) {
      float lf = (x - tg[j]) / (tg[j + k] - tg[j]);
      float rf = (tg[j + k + 1] - x) / (tg[j + k + 1] - tg[j + 1]);
      bb[j] = lf * bb[j] + rf * bb[j + 1];
    }
  }
  bf16_t* bo = outr + I + (size_t)i * 8;
#pragma unroll
  for (int k = 0; k < 8; ++k) bo[k] = f2bf(bb[k]);
}

// Pack [base_w | spline_w] into one augmented bf16 weight matrix (O, I*9).
__global__ void k_pack_kan_w(const float* __restrict__ bw,
                             const float* __restrict__ sw,
                             bf16_t* __restrict__ waug, int O, int I) {
  size_t idx = (size_t)blockIdx.x * blockDim.x + threadIdx.x;
  if (idx >= (size_t)O * I) return;
  int i = (int)(idx % I);
  int o = (int)(idx / I);
  bf16_t* row = waug + (size_t)o * (I * 9);
  row[i] = f2bf(bw[idx]);
  const float* sp = sw + idx * 8;
  bf16_t* dp = row + I + (size_t)i * 8;
#pragma unroll
  for (int k = 0; k < 8; ++k) dp[k] = f2bf(sp[k]);
}

// ---------------------------------------------------------------------------
// Orchestration
// ---------------------------------------------------------------------------
extern "C" void kernel_launch(void* const* d_in, const int* in_sizes, int n_in,
                              void* d_out, int out_size, void* d_ws,
                              size_t ws_size, hipStream_t stream) {
  const float* x       = (const float*)d_in[0];
  const float* terrain = (const float*)d_in[1];
  const float* qkv_w   = (const float*)d_in[2];
  const float* qkv_b   = (const float*)d_in[3];
  const float* out_w   = (const float*)d_in[4];
  const float* out_b   = (const float*)d_in[5];
  const float* ln1_s   = (const float*)d_in[6];
  const float* ln1_b   = (const float*)d_in[7];
  const float* bw0     = (const float*)d_in[8];
  const float* sw0     = (const float*)d_in[9];
  const float* bw1     = (const float*)d_in[10];
  const float* sw1     = (const float*)d_in[11];
  const float* ln2_s   = (const float*)d_in[12];
  const float* ln2_b   = (const float*)d_in[13];
  float* out = (float*)d_out;

  const int B = 2, S = 2048, E = 768, H = 12, D1 = 1536;
  const int N = B * S;        // 4096 rows
  const int E3 = 3 * E;       // 2304
  const int KA0 = E * 9;      // 6912
  const int KA1 = D1 * 9;     // 13824

  char* base = (char*)d_ws;

  // Arena for phase-1 temporaries; later fully reused by aug1 (113.3 MB).
  const size_t ARENA = (size_t)170 << 20;
  size_t aoff = 0;
  auto aalloc = [&](size_t bytes) -> void* {
    aoff = (aoff + 255) & ~(size_t)255;
    void* r = base + aoff;
    aoff += bytes;
    return r;
  };
  bf16_t* x_bf    = (bf16_t*)aalloc((size_t)N * E * 2);
  bf16_t* qkvw_bf = (bf16_t*)aalloc((size_t)E3 * E * 2);
  float*  qkv_f   = (float*)aalloc((size_t)N * E3 * 4);
  bf16_t* q_bf    = (bf16_t*)aalloc((size_t)N * E * 2);
  bf16_t* k_bf    = (bf16_t*)aalloc((size_t)N * E * 2);
  bf16_t* v_bf    = (bf16_t*)aalloc((size_t)N * E * 2);
  bf16_t* ao_bf   = (bf16_t*)aalloc((size_t)N * E * 2);
  bf16_t* outw_bf = (bf16_t*)aalloc((size_t)E * E * 2);
  float*  atn_f   = (float*)aalloc((size_t)N * E * 4);
  bf16_t* aug0    = (bf16_t*)aalloc((size_t)N * KA0 * 2);
  bf16_t* w0aug   = (bf16_t*)aalloc((size_t)D1 * KA0 * 2);
  bf16_t* aug1    = (bf16_t*)base;  // overlays entire arena (phase-1 is dead)

  size_t off = ARENA;
  auto palloc = [&](size_t bytes) -> void* {
    off = (off + 255) & ~(size_t)255;
    void* r = base + off;
    off += bytes;
    return r;
  };
  float*  h_f    = (float*)palloc((size_t)N * E * 4);
  float*  kan0_f = (float*)palloc((size_t)N * D1 * 4);
  bf16_t* w1aug  = (bf16_t*)palloc((size_t)E * KA1 * 2);
  float*  kan1_f = (float*)palloc((size_t)N * E * 4);

  const int thr = 256;
  auto blocks = [&](size_t n) -> unsigned {
    return (unsigned)((n + thr - 1) / thr);
  };

  // 1) bf16 conversions
  k_cvt_bf16<<<blocks((size_t)N * E), thr, 0, stream>>>(x, x_bf, (size_t)N * E);
  k_cvt_bf16<<<blocks((size_t)E3 * E), thr, 0, stream>>>(qkv_w, qkvw_bf,
                                                         (size_t)E3 * E);
  k_cvt_bf16<<<blocks((size_t)E * E), thr, 0, stream>>>(out_w, outw_bf,
                                                        (size_t)E * E);

  // 2) QKV projection: (4096 x 2304 x 768)
  k_gemm_bf16<<<dim3(E3 / 128, N / 128), 256, 0, stream>>>(
      x_bf, qkvw_bf, qkv_b, qkv_f, N, E3, E);

  // 3) split / transpose to (B,H,S,64) bf16
  k_split_qkv<<<blocks((size_t)N * E), thr, 0, stream>>>(qkv_f, q_bf, k_bf,
                                                         v_bf, B, H, S);

  // 4) flash attention -> ao bf16 in (B,S,E)
  k_attn<<<dim3(S / 64, H, B), 128, 0, stream>>>(q_bf, k_bf, v_bf, ao_bf, S, H);

  // 5) output projection: (4096 x 768 x 768)
  k_gemm_bf16<<<dim3(E / 128, N / 128), 256, 0, stream>>>(
      ao_bf, outw_bf, out_b, atn_f, N, E, E);

  // 6) h = LN1(x + attn) * sigmoid(terrain)
  k_add_ln<<<N, 256, 0, stream>>>(x, atn_f, ln1_s, ln1_b, terrain, h_f, E, S);

  // 7) KAN layer 0: aug = [silu(h)|bases(h)], one GEMM (4096x1536x6912)
  k_kan_prep<<<blocks((size_t)N * E), thr, 0, stream>>>(h_f, aug0, E, (size_t)N);
  k_pack_kan_w<<<blocks((size_t)D1 * E), thr, 0, stream>>>(bw0, sw0, w0aug, D1, E);
  k_gemm_bf16<<<dim3(D1 / 128, N / 128), 256, 0, stream>>>(
      aug0, w0aug, nullptr, kan0_f, N, D1, KA0);

  // 8) KAN layer 1: (4096x768x13824)
  k_kan_prep<<<blocks((size_t)N * D1), thr, 0, stream>>>(kan0_f, aug1, D1,
                                                         (size_t)N);
  k_pack_kan_w<<<blocks((size_t)E * D1), thr, 0, stream>>>(bw1, sw1, w1aug, E, D1);
  k_gemm_bf16<<<dim3(E / 128, N / 128), 256, 0, stream>>>(
      aug1, w1aug, nullptr, kan1_f, N, E, KA1);

  // 9) out = LN2(h + kan)
  k_add_ln<<<N, 256, 0, stream>>>(h_f, kan1_f, ln2_s, ln2_b, nullptr, out, E, S);
}
